// HybridInterferenceSelfAttention_37203006718037
// MI455X (gfx1250) — matrix operand order repaired and checked by reading, not compiled, not run on hardware
//
#include <hip/hip_runtime.h>
#include <math.h>

// ---------------------------------------------------------------------------
// Hybrid interference self-attention for MI455X (gfx1250, wave32, WMMA).
// B=2, S=2048, D=1024, H=16, HD=64. Extended attention feature dim = 192.
//
// Pipeline:
//   cvt:   x -> bf16 (once); each W -> bf16 transposed [n][k] (once)
//   gemm:  bf16 WMMA GEMM with double-buffered async global->LDS staging
//   prep:  softplus/RMS/phase features -> q_ext/k_ext (192) + V^T, all bf16
//   attn:  flash attention (12 WMMA scores over FE=192, 4 WMMA for P*V)
//   gemm:  O(bf16) @ Wo^T -> d_out (f32)
// ---------------------------------------------------------------------------

#define Bb 2
#define Ss 2048
#define Dd 1024
#define Hh 16
#define HDim 64
#define Mrows (Bb * Ss)         // 4096
#define FE 192                  // 64 (dot) + 64 (cos) + 64 (sin)

typedef __bf16 bf16;
typedef __attribute__((ext_vector_type(16))) bf16  v16bf;
typedef __attribute__((ext_vector_type(8)))  float v8f;
typedef int i32x4 __attribute__((vector_size(16)));

#if __has_builtin(__builtin_amdgcn_global_load_async_to_lds_b128)
#define ASYNC_LDS 1
typedef __attribute__((address_space(1))) i32x4 gi32x4;
typedef __attribute__((address_space(3))) i32x4 li32x4;
#endif

__device__ __forceinline__ v8f wmma_bf16(v16bf a, v16bf b, v8f c) {
  return __builtin_amdgcn_wmma_f32_16x16x32_bf16(
      false, a, false, b, (short)0, c, false, false);
}

__device__ __forceinline__ v16bf load16bf(const bf16* p) {
  union { uint4 u[2]; v16bf v; } t;
  t.u[0] = ((const uint4*)p)[0];
  t.u[1] = ((const uint4*)p)[1];
  return t.v;
}

__device__ __forceinline__ void load8bf(v16bf& d, int e0, const bf16* p) {
  union { uint4 u; bf16 h[8]; } t;
  t.u = *(const uint4*)p;
#pragma unroll
  for (int i = 0; i < 8; ++i) d[e0 + i] = t.h[i];
}

// 16B global -> LDS copy. Async (ASYNCcnt-tracked) when the CDNA5 builtin is
// available, else synchronous through VGPRs.
__device__ __forceinline__ void cp16_g2l(const bf16* g, bf16* l) {
#ifdef ASYNC_LDS
  __builtin_amdgcn_global_load_async_to_lds_b128(
      (gi32x4*)const_cast<bf16*>(g), (li32x4*)l, 0, 0);
#else
  *(uint4*)l = *(const uint4*)g;
#endif
}

__device__ __forceinline__ void wait_async_tail4() {
#ifdef ASYNC_LDS
  asm volatile("s_wait_asynccnt 0x4" ::: "memory");
#endif
}
__device__ __forceinline__ void wait_async_zero() {
#ifdef ASYNC_LDS
  asm volatile("s_wait_asynccnt 0x0" ::: "memory");
#endif
}

__device__ __forceinline__ float softplusf(float x) {
  return x > 15.0f ? x : log1pf(__expf(x));
}

// ---------------------------------------------------------------------------
// f32 -> bf16 bulk convert (8 elems/thread, 16B stores).
// ---------------------------------------------------------------------------
__global__ __launch_bounds__(256)
void cvt_bf16(const float* __restrict__ in, bf16* __restrict__ out) {
  size_t i = ((size_t)blockIdx.x * 256 + threadIdx.x) * 8;
  float4 a = *(const float4*)&in[i];
  float4 b = *(const float4*)&in[i + 4];
  union { uint4 u; bf16 h[8]; } o;
  o.h[0] = (bf16)a.x; o.h[1] = (bf16)a.y; o.h[2] = (bf16)a.z; o.h[3] = (bf16)a.w;
  o.h[4] = (bf16)b.x; o.h[5] = (bf16)b.y; o.h[6] = (bf16)b.z; o.h[7] = (bf16)b.w;
  *(uint4*)&out[i] = o.u;
}

// ---------------------------------------------------------------------------
// W[k][n] f32 -> Wt[n][k] bf16 (32x32 LDS tile transpose).
// ---------------------------------------------------------------------------
__global__ __launch_bounds__(256)
void cvt_wT(const float* __restrict__ W, bf16* __restrict__ Wt) {
  __shared__ float tile[32][33];
  const int tx = threadIdx.x, ty = threadIdx.y;
  const int kb = blockIdx.y * 32, nb = blockIdx.x * 32;
#pragma unroll
  for (int j = ty; j < 32; j += 8)
    tile[j][tx] = W[(size_t)(kb + j) * Dd + nb + tx];
  __syncthreads();
#pragma unroll
  for (int j = ty; j < 32; j += 8)
    Wt[(size_t)(nb + j) * Dd + kb + tx] = (bf16)tile[tx][j];
}

// ---------------------------------------------------------------------------
// GEMM: C[M,1024](f32) = A[M,1024](bf16, row-major) * Wt[1024,1024](bf16,
// [n][k]). Block = 128 threads / 4 waves, 64x64 tile, K-step 32.
// Double-buffered async global->LDS staging gated by s_wait_asynccnt.
// ---------------------------------------------------------------------------
__global__ __launch_bounds__(128)
void gemm_bf(const bf16* __restrict__ A, const bf16* __restrict__ Wt,
             float* __restrict__ C) {
  __shared__ alignas(16) bf16 Ab[2][64 * 32];   // [row][k]
  __shared__ alignas(16) bf16 Bt[2][64 * 32];   // [n][k]

  const int m0   = blockIdx.y * 64;
  const int n0   = blockIdx.x * 64;
  const int t    = threadIdx.x;
  const int w    = t >> 5;
  const int lane = t & 31;
  const int half = lane >> 4;
  const int lm   = lane & 15;

  // Per-thread staging map: 2 chunks of 16B each for A and for B per stage.
  const int c0 = t, c1 = t + 128;               // chunk ids 0..255
  const int r0 = c0 >> 2, s0c = (c0 & 3) * 8;   // row, k-offset (bf16 elems)
  const int r1 = c1 >> 2, s1c = (c1 & 3) * 8;

  v8f acc[4];
#pragma unroll
  for (int nt = 0; nt < 4; ++nt)
#pragma unroll
    for (int v = 0; v < 8; ++v) acc[nt][v] = 0.0f;

  // issue stage for buf,kb  (4 async 16B copies per thread)
  auto stage = [&](int buf, int kb) {
    cp16_g2l(&A [(size_t)(m0 + r0) * Dd + kb + s0c], &Ab[buf][r0 * 32 + s0c]);
    cp16_g2l(&A [(size_t)(m0 + r1) * Dd + kb + s1c], &Ab[buf][r1 * 32 + s1c]);
    cp16_g2l(&Wt[(size_t)(n0 + r0) * Dd + kb + s0c], &Bt[buf][r0 * 32 + s0c]);
    cp16_g2l(&Wt[(size_t)(n0 + r1) * Dd + kb + s1c], &Bt[buf][r1 * 32 + s1c]);
  };

  stage(0, 0);
  int buf = 0;
  for (int kb = 0; kb < Dd; kb += 32, buf ^= 1) {
    if (kb + 32 < Dd) {
      stage(buf ^ 1, kb + 32);
      wait_async_tail4();        // oldest 4 (this buf) done; next 4 in flight
    } else {
      wait_async_zero();
    }
    __syncthreads();

    v16bf af;
    const bf16* ar = &Ab[buf][(w * 16 + lm) * 32];
    load8bf(af, 0, ar + half * 16);
    load8bf(af, 8, ar + half * 16 + 8);
#pragma unroll
    for (int nt = 0; nt < 4; ++nt) {
      v16bf bfg = load16bf(&Bt[buf][(nt * 16 + lm) * 32 + half * 16]);
      acc[nt] = wmma_bf16(af, bfg, acc[nt]);
    }
    __syncthreads();             // all waves done with buf before restaging
  }

#pragma unroll
  for (int nt = 0; nt < 4; ++nt)
#pragma unroll
    for (int v = 0; v < 8; ++v) {
      int row = m0 + w * 16 + v + half * 8;
      int col = n0 + nt * 16 + lm;
      C[(size_t)row * Dd + col] = acc[nt][v];
    }
}

// ---------------------------------------------------------------------------
// Prep: per (b,s,h) row build bf16 extended features + transposed V.
// One wave per row; lane owns 2 of the 64 head dims.
// ---------------------------------------------------------------------------
__global__ __launch_bounds__(128)
void prep(const float* __restrict__ Q, const float* __restrict__ K,
          const float* __restrict__ Qp, const float* __restrict__ Kp,
          const float* __restrict__ V, const float* __restrict__ lambda_raw,
          bf16* __restrict__ qext, bf16* __restrict__ kext,
          bf16* __restrict__ vt) {
  const int wid   = threadIdx.x >> 5;
  const int lane  = threadIdx.x & 31;
  const int rowid = blockIdx.x * 4 + wid;       // 0 .. B*S*H-1
  const int bs = rowid >> 4;
  const int h  = rowid & 15;
  const int b  = bs / Ss;
  const int s  = bs % Ss;
  const int hd0 = lane * 2;

  const size_t base = (size_t)bs * Dd + h * HDim;
  float q0 = Q[base + hd0],  q1 = Q[base + hd0 + 1];
  float k0 = K[base + hd0],  k1 = K[base + hd0 + 1];
  float qp0 = Qp[base + hd0], qp1 = Qp[base + hd0 + 1];
  float kp0 = Kp[base + hd0], kp1 = Kp[base + hd0 + 1];
  float v0 = V[base + hd0],  v1 = V[base + hd0 + 1];

  float aq0 = softplusf(q0), aq1 = softplusf(q1);
  float ak0 = softplusf(k0), ak1 = softplusf(k1);
  float sq = aq0 * aq0 + aq1 * aq1;
  float sk = ak0 * ak0 + ak1 * ak1;
#pragma unroll
  for (int off = 16; off > 0; off >>= 1) {
    sq += __shfl_xor(sq, off, 32);
    sk += __shfl_xor(sk, off, 32);
  }
  float rq = rsqrtf(sq * (1.0f / 64.0f) + 1e-6f);
  float rk = rsqrtf(sk * (1.0f / 64.0f) + 1e-6f);
  aq0 *= rq; aq1 *= rq; ak0 *= rk; ak1 *= rk;

  const float PI = 3.14159265358979323846f;
  float phq0 = PI * tanhf(qp0), phq1 = PI * tanhf(qp1);
  float phk0 = PI * tanhf(kp0), phk1 = PI * tanhf(kp1);

  const float lam   = 0.5f * tanhf(lambda_raw[h]);
  const float scale = 0.125f;                   // 1/sqrt(64)
  const float sl    = scale * lam;

  const size_t eb = ((size_t)(b * Hh + h) * Ss + s) * FE;
  qext[eb + hd0]           = (bf16)(scale * q0);
  qext[eb + hd0 + 1]       = (bf16)(scale * q1);
  qext[eb + 64 + hd0]      = (bf16)(sl * aq0 * __cosf(phq0));
  qext[eb + 64 + hd0 + 1]  = (bf16)(sl * aq1 * __cosf(phq1));
  qext[eb + 128 + hd0]     = (bf16)(sl * aq0 * __sinf(phq0));
  qext[eb + 128 + hd0 + 1] = (bf16)(sl * aq1 * __sinf(phq1));

  kext[eb + hd0]           = (bf16)k0;
  kext[eb + hd0 + 1]       = (bf16)k1;
  kext[eb + 64 + hd0]      = (bf16)(ak0 * __cosf(phk0));
  kext[eb + 64 + hd0 + 1]  = (bf16)(ak1 * __cosf(phk1));
  kext[eb + 128 + hd0]     = (bf16)(ak0 * __sinf(phk0));
  kext[eb + 128 + hd0 + 1] = (bf16)(ak1 * __sinf(phk1));

  const size_t vb = ((size_t)(b * Hh + h) * HDim + hd0) * Ss + s;
  vt[vb]      = (bf16)v0;
  vt[vb + Ss] = (bf16)v1;
}

// ---------------------------------------------------------------------------
// Flash attention: one wave per (b, h, 16-query block); keys 32 at a time.
// Output written directly as bf16 for the final GEMM.
// ---------------------------------------------------------------------------
__global__ __launch_bounds__(128)
void attn(const bf16* __restrict__ qext, const bf16* __restrict__ kext,
          const bf16* __restrict__ vt, bf16* __restrict__ Ob) {
  __shared__ alignas(16) float Pt[4][16][32];   // per-wave P transpose tile

  const int wid  = threadIdx.x >> 5;
  const int lane = threadIdx.x & 31;
  const int half = lane >> 4;
  const int lm   = lane & 15;
  const int qi   = blockIdx.x * 4 + wid;        // 0..127
  const int h    = blockIdx.y;
  const int b    = blockIdx.z;
  const int bh   = b * Hh + h;

  v16bf qa[6];
  {
    const bf16* qrow = qext + ((size_t)bh * Ss + qi * 16 + lm) * FE;
#pragma unroll
    for (int fc = 0; fc < 6; ++fc) {
      const bf16* p = qrow + fc * 32 + half * 16;
      load8bf(qa[fc], 0, p);
      load8bf(qa[fc], 8, p + 8);
    }
  }

  v8f o[4];
#pragma unroll
  for (int nt = 0; nt < 4; ++nt)
#pragma unroll
    for (int v = 0; v < 8; ++v) o[nt][v] = 0.0f;
  float m[8], l[8];
#pragma unroll
  for (int v = 0; v < 8; ++v) { m[v] = -INFINITY; l[v] = 0.0f; }

  const int nch = (qi + 2) / 2;

  for (int c = 0; c < nch; ++c) {
    const int j0 = c * 32;

    v8f s0, s1;
#pragma unroll
    for (int v = 0; v < 8; ++v) { s0[v] = 0.0f; s1[v] = 0.0f; }
#pragma unroll
    for (int fc = 0; fc < 6; ++fc) {
      const bf16* k0p =
          kext + ((size_t)bh * Ss + j0 + lm) * FE + fc * 32 + half * 16;
      const bf16* k1p =
          kext + ((size_t)bh * Ss + j0 + 16 + lm) * FE + fc * 32 + half * 16;
      v16bf b0 = load16bf(k0p);
      v16bf b1 = load16bf(k1p);
      s0 = wmma_bf16(qa[fc], b0, s0);
      s1 = wmma_bf16(qa[fc], b1, s1);
    }

    if (c == nch - 1) {
#pragma unroll
      for (int v = 0; v < 8; ++v) {
        int i = qi * 16 + v + half * 8;
        if (j0 + lm > i)      s0[v] = -INFINITY;
        if (j0 + 16 + lm > i) s1[v] = -INFINITY;
      }
    }

#pragma unroll
    for (int v = 0; v < 8; ++v) {
      float tmax = fmaxf(s0[v], s1[v]);
#pragma unroll
      for (int off = 8; off > 0; off >>= 1)
        tmax = fmaxf(tmax, __shfl_xor(tmax, off, 16));
      float mn = fmaxf(m[v], tmax);
      float alpha = __expf(m[v] - mn);
      m[v] = mn;
      float p0 = __expf(s0[v] - mn);
      float p1 = __expf(s1[v] - mn);
      float r = p0 + p1;
#pragma unroll
      for (int off = 8; off > 0; off >>= 1) r += __shfl_xor(r, off, 16);
      l[v] = l[v] * alpha + r;
#pragma unroll
      for (int nt = 0; nt < 4; ++nt) o[nt][v] *= alpha;
      Pt[wid][v + half * 8][lm]      = p0;
      Pt[wid][v + half * 8][16 + lm] = p1;
    }
    // Wave-private LDS tile: hardware services DS in order; the wait +
    // memory clobber stops the compiler reordering cross-lane traffic.
    asm volatile("s_wait_dscnt 0" ::: "memory");

    v16bf pa;
#pragma unroll
    for (int e = 0; e < 8; ++e)
      pa[e] = (bf16)Pt[wid][lm][half * 16 + e];
#pragma unroll
    for (int e = 0; e < 8; ++e)
      pa[8 + e] = (bf16)Pt[wid][lm][half * 16 + 8 + e];
    asm volatile("s_wait_dscnt 0" ::: "memory");

#pragma unroll
    for (int nt = 0; nt < 4; ++nt) {
      const bf16* vp =
          vt + ((size_t)bh * HDim + nt * 16 + lm) * Ss + j0 + half * 16;
      v16bf vb = load16bf(vp);
      o[nt] = wmma_bf16(pa, vb, o[nt]);
    }
  }

#pragma unroll
  for (int nt = 0; nt < 4; ++nt)
#pragma unroll
    for (int v = 0; v < 8; ++v) {
      int row = qi * 16 + v + half * 8;
      int col = h * HDim + nt * 16 + lm;
      Ob[((size_t)b * Ss + row) * Dd + col] = (bf16)(o[nt][v] / l[v]);
    }
}

// ---------------------------------------------------------------------------
// Launcher. Inputs: x, Wq, Wk, Wv, Wo, Wqp, Wkp, lambda_raw (all f32).
// ---------------------------------------------------------------------------
extern "C" void kernel_launch(void* const* d_in, const int* in_sizes, int n_in,
                              void* d_out, int out_size, void* d_ws,
                              size_t ws_size, hipStream_t stream) {
  const float* x   = (const float*)d_in[0];
  const float* Wf[6] = {(const float*)d_in[1], (const float*)d_in[2],
                        (const float*)d_in[3], (const float*)d_in[5],
                        (const float*)d_in[6], (const float*)d_in[4]};
  // order: Wq, Wk, Wv, Wqp, Wkp, Wo
  const float* lr = (const float*)d_in[7];

  const size_t NE = (size_t)Mrows * Dd;           // 4M elems
  const size_t WE = (size_t)Dd * Dd;              // 1M elems
  float* Q  = (float*)d_ws;
  float* K  = Q + NE;
  float* V  = K + NE;
  float* Qp = V + NE;
  float* Kp = Qp + NE;
  bf16* qext = (bf16*)(Kp + NE);
  bf16* kext = qext + (size_t)Bb * Hh * Ss * FE;
  bf16* vtb  = kext + (size_t)Bb * Hh * Ss * FE;
  bf16* xb   = vtb + (size_t)Bb * Hh * HDim * Ss;
  bf16* Ob   = xb + NE;
  bf16* Wt[6];
  Wt[0] = Ob + NE;
  for (int i = 1; i < 6; ++i) Wt[i] = Wt[i - 1] + WE;

  // One-time bf16 conversions (x row-major; weights transposed to [n][k]).
  cvt_bf16<<<NE / (256 * 8), 256, 0, stream>>>(x, xb);
  dim3 tb(32, 8), tg(Dd / 32, Dd / 32);
  for (int i = 0; i < 6; ++i) cvt_wT<<<tg, tb, 0, stream>>>(Wf[i], Wt[i]);

  dim3 gg(Dd / 64, Mrows / 64);
  gemm_bf<<<gg, 128, 0, stream>>>(xb, Wt[0], Q);
  gemm_bf<<<gg, 128, 0, stream>>>(xb, Wt[1], K);
  gemm_bf<<<gg, 128, 0, stream>>>(xb, Wt[2], V);
  gemm_bf<<<gg, 128, 0, stream>>>(xb, Wt[3], Qp);
  gemm_bf<<<gg, 128, 0, stream>>>(xb, Wt[4], Kp);

  prep<<<(Bb * Ss * Hh) / 4, 128, 0, stream>>>(Q, K, Qp, Kp, V, lr,
                                               qext, kext, vtb);

  attn<<<dim3(Ss / 16 / 4, Hh, Bb), 128, 0, stream>>>(qext, kext, vtb, Ob);

  gemm_bf<<<gg, 128, 0, stream>>>(Ob, Wt[5], (float*)d_out);
}
